// InfoNCE_17231408791669
// MI455X (gfx1250) — matrix operand order repaired and checked by reading, not compiled, not run on hardware
//
#include <hip/hip_runtime.h>

typedef __attribute__((ext_vector_type(16))) _Float16 v16h;
typedef __attribute__((ext_vector_type(8)))  _Float16 v8h;
typedef __attribute__((ext_vector_type(8)))  float    v8f;
typedef __attribute__((ext_vector_type(4)))  float    v4f;

#define NB      8192        // batch (rows of both features)
#define DIMK    512         // feature dim
#define TEMP_INV (1.0f / 0.07f)

#define TM 256              // block tile M
#define TN 128              // block tile N
#define TK 32               // K per WMMA step
#define LDSS 40             // padded LDS row stride in halfs (conflict-free b128)

// ---------------- Pass 0: row-normalize f32 -> f16 (torch eps clamp) ----------------
__global__ __launch_bounds__(256)
void normalize_f16_kernel(const float* __restrict__ in, _Float16* __restrict__ out) {
  const int wave = threadIdx.x >> 5;
  const int lane = threadIdx.x & 31;
  const int row  = blockIdx.x * 8 + wave;      // one wave per row
  const float* src = in + (size_t)row * DIMK + lane * 16;

  v4f x0 = *(const v4f*)(src + 0);
  v4f x1 = *(const v4f*)(src + 4);
  v4f x2 = *(const v4f*)(src + 8);
  v4f x3 = *(const v4f*)(src + 12);

  float ss = 0.f;
#pragma unroll
  for (int i = 0; i < 4; ++i)
    ss += x0[i]*x0[i] + x1[i]*x1[i] + x2[i]*x2[i] + x3[i]*x3[i];
#pragma unroll
  for (int m = 16; m >= 1; m >>= 1) ss += __shfl_xor(ss, m, 32);

  const float scale = 1.0f / fmaxf(sqrtf(ss), 1e-8f);

  v8h h0, h1;
#pragma unroll
  for (int i = 0; i < 4; ++i) {
    h0[i]     = (_Float16)(x0[i] * scale);
    h0[i + 4] = (_Float16)(x1[i] * scale);
    h1[i]     = (_Float16)(x2[i] * scale);
    h1[i + 4] = (_Float16)(x3[i] * scale);
  }
  _Float16* dst = out + (size_t)row * DIMK + lane * 16;
  *(v8h*)dst       = h0;
  *(v8h*)(dst + 8) = h1;
}

// ---------------- Pass 1: cos = A * B^T via v_wmma_f32_16x16x32_f16 ----------------
// Block: 256 threads (8 waves), tile 256x128. Wave: 64x64 (4x4 WMMA tiles).
__global__ __launch_bounds__(256)
void wmma_gemm_kernel(const _Float16* __restrict__ A,   // [NB][DIMK] normalized a
                      const _Float16* __restrict__ Bm,  // [NB][DIMK] normalized b
                      float* __restrict__ C) {          // [NB][NB]
  __shared__ _Float16 As[TM * LDSS];   // 256 x 40 halfs = 20.0 KB
  __shared__ _Float16 Bs[TN * LDSS];   // 128 x 40 halfs = 10.0 KB

  const int mBase = blockIdx.y * TM;
  const int nBase = blockIdx.x * TN;
  const int t    = threadIdx.x;
  const int lane = t & 31;
  const int wave = t >> 5;
  const int mW = (wave >> 1) * 64;     // 4 wave-rows x 64
  const int nW = (wave & 1) * 64;      // 2 wave-cols x 64

  // global->LDS mapping: 4 lanes cover 64 contiguous bytes of one row
  const int seg = t & 3;               // k segment: halfs [seg*8, seg*8+8)
  const int r0  = t >> 2;              // base tile row 0..63

  const _Float16* aPtr = A  + (size_t)(mBase + r0) * DIMK + seg * 8;
  const _Float16* bPtr = Bm + (size_t)(nBase + r0) * DIMK + seg * 8;

  v8f acc[4][4];
  v8f zero = {};
#pragma unroll
  for (int mt = 0; mt < 4; ++mt)
#pragma unroll
    for (int nt = 0; nt < 4; ++nt) acc[mt][nt] = zero;

  const int am  = lane & 15;
  const int akb = (lane >> 4) * 16;    // A frag: K base 0/16
  const int bn  = lane & 15;
  const int bkb = (lane >> 4) * 16;    // B frag: K base 0/16

  for (int k0 = 0; k0 < DIMK; k0 += TK) {
    // stage next tile through registers, then LDS
    v8h aw[4], bw[2];
#pragma unroll
    for (int i = 0; i < 4; ++i)
      aw[i] = *(const v8h*)(aPtr + (size_t)(64 * i) * DIMK + k0);
#pragma unroll
    for (int i = 0; i < 2; ++i)
      bw[i] = *(const v8h*)(bPtr + (size_t)(64 * i) * DIMK + k0);

    __syncthreads();
#pragma unroll
    for (int i = 0; i < 4; ++i)
      *(v8h*)(&As[(r0 + 64 * i) * LDSS + seg * 8]) = aw[i];
#pragma unroll
    for (int i = 0; i < 2; ++i)
      *(v8h*)(&Bs[(r0 + 64 * i) * LDSS + seg * 8]) = bw[i];
    __syncthreads();

    // A fragments: lane holds M=am, K = akb+{0..15} (consecutive k of row am)
    v16h afrag[4];
#pragma unroll
    for (int mt = 0; mt < 4; ++mt) {
      const _Float16* p = &As[(mW + mt * 16 + am) * LDSS + akb];
      v8h lo = *(const v8h*)p;
      v8h hi = *(const v8h*)(p + 8);
#pragma unroll
      for (int i = 0; i < 8; ++i) { afrag[mt][i] = lo[i]; afrag[mt][8 + i] = hi[i]; }
    }
    // B fragments: lane holds N=bn, K = bkb+{0..15} (consecutive k of row bn)
    v16h bfrag[4];
#pragma unroll
    for (int nt = 0; nt < 4; ++nt) {
      const _Float16* p = &Bs[(nW + nt * 16 + bn) * LDSS + bkb];
      v8h lo = *(const v8h*)p;
      v8h hi = *(const v8h*)(p + 8);
#pragma unroll
      for (int i = 0; i < 8; ++i) { bfrag[nt][i] = lo[i]; bfrag[nt][8 + i] = hi[i]; }
    }

#pragma unroll
    for (int mt = 0; mt < 4; ++mt)
#pragma unroll
      for (int nt = 0; nt < 4; ++nt)
        acc[mt][nt] = __builtin_amdgcn_wmma_f32_16x16x32_f16(
            false, afrag[mt], false, bfrag[nt],
            (short)0, acc[mt][nt], false, false);
  }

  // store: D layout — lane -> N=lane&15, VGPR r -> M = r + (lane>>4)*8
  const int col0  = nBase + nW + (lane & 15);
  const int cmOff = (lane >> 4) * 8;
#pragma unroll
  for (int mt = 0; mt < 4; ++mt)
#pragma unroll
    for (int nt = 0; nt < 4; ++nt)
#pragma unroll
      for (int r = 0; r < 8; ++r) {
        const int row = mBase + mW + mt * 16 + cmOff + r;
        C[(size_t)row * NB + col0 + nt * 16] = acc[mt][nt][r];
      }
}

// ---------------- Pass 2: per-row masked softmax sums -> log(den)-log(num) ----------------
__global__ __launch_bounds__(256)
void row_loss_kernel(const float* __restrict__ cosm, const int* __restrict__ labels,
                     float* __restrict__ lossRow) {
  __shared__ int lbl[NB];          // 32 KB: cache all labels once per block
  const int t = threadIdx.x;
  for (int j = t; j < NB; j += 256) lbl[j] = labels[j];
  __syncthreads();

  const int row = blockIdx.x;
  const int myLbl = lbl[row];
  const float* cr = cosm + (size_t)row * NB;

  float num = 0.f, den = 0.f;
#pragma unroll 4
  for (int j = 0; j < NB; j += 256) {
    const float e = __expf(cr[j + t] * TEMP_INV);
    den += e;
    if (lbl[j + t] == myLbl) num += e;
  }
#pragma unroll
  for (int m = 16; m >= 1; m >>= 1) {
    num += __shfl_xor(num, m, 32);
    den += __shfl_xor(den, m, 32);
  }
  __shared__ float rn[8], rd[8];
  const int wave = t >> 5, lane = t & 31;
  if (lane == 0) { rn[wave] = num; rd[wave] = den; }
  __syncthreads();
  if (t == 0) {
    float N = 0.f, D = 0.f;
#pragma unroll
    for (int w = 0; w < 8; ++w) { N += rn[w]; D += rd[w]; }
    lossRow[row] = logf(D) - logf(N);    // = -log(num/den)
  }
}

// ---------------- Pass 3: deterministic mean over rows ----------------
__global__ __launch_bounds__(256)
void final_reduce_kernel(const float* __restrict__ lossRow, float* __restrict__ out) {
  const int t = threadIdx.x;
  float s = 0.f;
  for (int j = t; j < NB; j += 256) s += lossRow[j];
#pragma unroll
  for (int m = 16; m >= 1; m >>= 1) s += __shfl_xor(s, m, 32);
  __shared__ float r[8];
  const int wave = t >> 5, lane = t & 31;
  if (lane == 0) r[wave] = s;
  __syncthreads();
  if (t == 0) {
    float tot = 0.f;
#pragma unroll
    for (int w = 0; w < 8; ++w) tot += r[w];
    out[0] = tot / (float)NB;
  }
}

extern "C" void kernel_launch(void* const* d_in, const int* in_sizes, int n_in,
                              void* d_out, int out_size, void* d_ws, size_t ws_size,
                              hipStream_t stream) {
  (void)in_sizes; (void)n_in; (void)out_size; (void)ws_size;

  const int*   labels = (const int*)d_in[0];   // label_tensor (int32 on device)
  const float* fa     = (const float*)d_in[1]; // feature_a [8192,512] f32
  const float* fb     = (const float*)d_in[2]; // feature_b [8192,512] f32

  float* out  = (float*)d_out;
  float* loss = out;       // d_out[0] = scalar loss
  float* cosm = out + 1;   // d_out[1..] = cos_score [8192,8192]

  _Float16* aH = (_Float16*)d_ws;                         // 8 MB
  _Float16* bH = aH + (size_t)NB * DIMK;                  // 8 MB
  float* lossRow = (float*)(bH + (size_t)NB * DIMK);      // 32 KB

  normalize_f16_kernel<<<NB / 8, 256, 0, stream>>>(fa, aH);
  normalize_f16_kernel<<<NB / 8, 256, 0, stream>>>(fb, bH);

  dim3 grid(NB / TN, NB / TM);
  wmma_gemm_kernel<<<grid, 256, 0, stream>>>(aH, bH, cosm);

  row_loss_kernel<<<NB, 256, 0, stream>>>(cosm, labels, lossRow);
  final_reduce_kernel<<<1, 256, 0, stream>>>(lossRow, loss);
}
